// PointNet2Encoder_43782896615853
// MI455X (gfx1250) — matrix operand order repaired and checked by reading, not compiled, not run on hardware
//
#include <hip/hip_runtime.h>
#include <hip/hip_bf16.h>

// ---------------------------------------------------------------------------
// PointNet++ encoder for MI455X (gfx1250, wave32, WMMA).
// Activations kept bf16 end-to-end with K padded to 64; GEMMs use
// v_wmma_f32_16x16x32_bf16. Each wave computes a 32x32 output tile
// (4 accumulators, 2x2 fragment reuse) -> 8 WMMAs per 64-k iteration.
// LDS tiles stored in fragment order so operand loads are ds_load_b128.
// ---------------------------------------------------------------------------

typedef __bf16 bf16_t;
typedef __bf16 bf16x4  __attribute__((ext_vector_type(4)));
typedef __bf16 bf16x16 __attribute__((ext_vector_type(16)));
typedef float  f32x8   __attribute__((ext_vector_type(8)));

static constexpr int   BATCH  = 8;
static constexpr int   NPTS   = 8192;
static constexpr float BN_EPS = 1e-5f;

#define DEVFN __device__ __forceinline__

DEVFN bf16_t f2bf(float x) { return (bf16_t)x; }

// -------------------------- (B,3,n) -> (B,n,3) ------------------------------
__global__ __launch_bounds__(256) void k_transpose_b3n(
    const float* __restrict__ in, float* __restrict__ out, int n) {
  int i = blockIdx.x * blockDim.x + threadIdx.x;
  if (i >= BATCH * n) return;
  int b = i / n, p = i % n;
  const float* src = in + (size_t)b * 3 * n + p;
  float* dst = out + ((size_t)b * n + p) * 3;
  dst[0] = src[0 * n];
  dst[1] = src[1 * n];
  dst[2] = src[2 * n];
}

// -------------------------- farthest point sampling --------------------------
__global__ __launch_bounds__(256) void k_fps(
    const float* __restrict__ xyz, int n, int npoint, int* __restrict__ idx) {
  const int b = blockIdx.x;
  const int t = threadIdx.x;
  const float* X = xyz + (size_t)b * n * 3;
  __shared__ float rs[256];
  __shared__ int   ri[256];
  __shared__ int   s_far;
  if (t == 0) s_far = 0;
  __syncthreads();

  float dist[32];                         // n <= 8192 -> <= 32 points/thread
  const int per = (n + 255) >> 8;
  for (int j = 0; j < per; ++j) dist[j] = 1e10f;

  for (int s = 0; s < npoint; ++s) {
    int far = s_far;
    if (t == 0) idx[b * npoint + s] = far;      // record BEFORE update
    float cx = X[far * 3 + 0], cy = X[far * 3 + 1], cz = X[far * 3 + 2];
    float best = -1.0f; int besti = 0;
    for (int j = 0; j < per; ++j) {
      int p = t + (j << 8);
      if (p < n) {
        float dx = X[p * 3 + 0] - cx;
        float dy = X[p * 3 + 1] - cy;
        float dz = X[p * 3 + 2] - cz;
        float d = dx * dx + dy * dy + dz * dz;
        float dm = fminf(dist[j], d);
        dist[j] = dm;
        if (dm > best) { best = dm; besti = p; }
      }
    }
    rs[t] = best; ri[t] = besti;
    __syncthreads();
    for (int o = 128; o > 0; o >>= 1) {
      if (t < o) {
        if (rs[t + o] > rs[t] || (rs[t + o] == rs[t] && ri[t + o] < ri[t])) {
          rs[t] = rs[t + o]; ri[t] = ri[t + o];
        }
      }
      __syncthreads();
    }
    if (t == 0) s_far = ri[0];
    __syncthreads();
  }
}

// -------------------------- gather sampled centers ---------------------------
__global__ __launch_bounds__(256) void k_gather_xyz(
    const float* __restrict__ xyz, const int* __restrict__ fpsidx,
    int n, int S, float* __restrict__ newxyz) {
  int i = blockIdx.x * blockDim.x + threadIdx.x;
  if (i >= BATCH * S) return;
  int b = i / S, s = i % S;
  int p = fpsidx[b * S + s];
  const float* src = xyz + ((size_t)b * n + p) * 3;
  float* dst = newxyz + (size_t)i * 3;
  dst[0] = src[0]; dst[1] = src[1]; dst[2] = src[2];
}

// -------------------------- ball query ---------------------------------------
__global__ __launch_bounds__(256) void k_ball(
    const float* __restrict__ xyz, const float* __restrict__ newxyz,
    int n, int S, int K, float r2, int* __restrict__ idx) {
  int i = blockIdx.x * blockDim.x + threadIdx.x;
  if (i >= BATCH * S) return;
  int b = i / S;
  const float* X = xyz + (size_t)b * n * 3;
  float cx = newxyz[(size_t)i * 3 + 0];
  float cy = newxyz[(size_t)i * 3 + 1];
  float cz = newxyz[(size_t)i * 3 + 2];
  int* out = idx + (size_t)i * K;
  int cnt = 0, first = 0;
  for (int p = 0; p < n && cnt < K; ++p) {
    float dx = X[p * 3 + 0] - cx;
    float dy = X[p * 3 + 1] - cy;
    float dz = X[p * 3 + 2] - cz;
    if (dx * dx + dy * dy + dz * dz <= r2) {
      if (cnt == 0) first = p;
      out[cnt++] = p;
    }
  }
  for (int c = cnt; c < K; ++c) out[c] = first;
}

// -------------------------- grouping (writes padded bf16 rows) ---------------
__global__ __launch_bounds__(256) void k_group_bf(
    const float* __restrict__ xyz, const float* __restrict__ newxyz,
    const float* __restrict__ feats, int n, int S, int K, int Cprev, int Cld,
    const int* __restrict__ idx, bf16_t* __restrict__ out) {
  int row = blockIdx.x * blockDim.x + threadIdx.x;
  if (row >= BATCH * S * K) return;
  int b = row / (S * K);
  int s = (row / K) % S;
  int p = idx[row];
  bf16_t* o = out + (size_t)row * Cld;
  const float* xp = xyz + ((size_t)b * n + p) * 3;
  const float* nx = newxyz + ((size_t)b * S + s) * 3;
  o[0] = f2bf(xp[0] - nx[0]);
  o[1] = f2bf(xp[1] - nx[1]);
  o[2] = f2bf(xp[2] - nx[2]);
  const float* f = feats + ((size_t)b * n + p) * Cprev;
  for (int c = 0; c < Cprev; ++c) o[3 + c] = f2bf(f[c]);
  for (int c = 3 + Cprev; c < Cld; ++c) o[c] = f2bf(0.0f);
}

// Stage 3 (group_all): rows = B*n, concat [xyz | feats], zero-pad to Cld.
__global__ __launch_bounds__(256) void k_group_all_bf(
    const float* __restrict__ xyz, const float* __restrict__ feats,
    int n, int Cprev, int Cld, bf16_t* __restrict__ out) {
  int row = blockIdx.x * blockDim.x + threadIdx.x;
  if (row >= BATCH * n) return;
  bf16_t* o = out + (size_t)row * Cld;
  const float* xp = xyz + (size_t)row * 3;
  o[0] = f2bf(xp[0]); o[1] = f2bf(xp[1]); o[2] = f2bf(xp[2]);
  const float* f = feats + (size_t)row * Cprev;
  for (int c = 0; c < Cprev; ++c) o[3 + c] = f2bf(f[c]);
  for (int c = 3 + Cprev; c < Cld; ++c) o[c] = f2bf(0.0f);
}

// -------------------------- weight packing: f32 (Nc,Kc) -> bf16 (Nc,Kld) -----
__global__ __launch_bounds__(256) void k_pack_w(
    const float* __restrict__ W, int Nc, int Kc, int Kld,
    bf16_t* __restrict__ Wb) {
  int i = blockIdx.x * blockDim.x + threadIdx.x;
  if (i >= Nc * Kld) return;
  int nn = i / Kld, k = i % Kld;
  Wb[i] = (k < Kc) ? f2bf(W[(size_t)nn * Kc + k]) : f2bf(0.0f);
}

// -------------------------- WMMA GEMM + bias ---------------------------------
// C[M,Nc] = A[M,Kld] * Wb^T + bias ; A,Wb bf16 with padded Kld (mult of 64).
// Block: 256 threads = 8 waves; 128x64 block tile; wave = 32x32 output
// (4 accumulators, 2x2 fragment reuse). K-step 64 -> 8 WMMAs per iteration.
// LDS tiles stored in fragment order: each lane's operand = 32 contiguous B.
__global__ __launch_bounds__(256) void k_gemm_bf16(
    const bf16_t* __restrict__ A, const bf16_t* __restrict__ Wb,
    const float* __restrict__ bias, float* __restrict__ C,
    int M, int Kld, int Nc) {
  __shared__ bf16_t sA[128 * 64];  // [m][k]        16 KB
  __shared__ bf16_t sB[64 * 64];   // [n][k]        8 KB
  const int tid  = threadIdx.x;
  const int lane = tid & 31;
  const int wave = tid >> 5;
  const int wm   = wave & 3;       // 4 wave-rows of 32 rows (2 m-tiles each)
  const int wn   = wave >> 2;      // 2 wave-cols of 32 cols (2 n-tiles each)
  const int m0   = blockIdx.x * 128;
  const int n0   = blockIdx.y * 64;
  const int half = lane >> 4;
  const int l15  = lane & 15;

  f32x8 acc00 = {}, acc01 = {}, acc10 = {}, acc11 = {};

  for (int k0 = 0; k0 < Kld; k0 += 64) {
    if (k0 + 64 < Kld)
      __builtin_prefetch(A + (size_t)(m0 + (tid >> 1)) * Kld + k0 + 64, 0, 1);

    // Stage A: 128 rows x 16 groups-of-4 = 2048 groups; 8 per thread.
    // k stored unpermuted (contiguous order, same as B).
#pragma unroll
    for (int j = 0; j < 8; ++j) {
      int g   = tid + 256 * j;
      int row = g >> 4;                       // 0..127
      int kg  = (g & 15) << 2;                // 0,4,...,60
      bf16x4 av = *(const bf16x4*)(A + (size_t)(m0 + row) * Kld + k0 + kg);
      int k5  = kg & 31;
      int pos = (kg & 32) + k5;
      *(bf16x4*)(sA + row * 64 + pos) = av;
    }
    // Stage B: 64 out-channels x 16 groups = 1024 groups; 4 per thread.
#pragma unroll
    for (int j = 0; j < 4; ++j) {
      int g   = tid + 256 * j;
      int row = g >> 4;                       // 0..63
      int kg  = (g & 15) << 2;
      bf16x4 bv = *(const bf16x4*)(Wb + (size_t)(n0 + row) * Kld + k0 + kg);
      *(bf16x4*)(sB + row * 64 + kg) = bv;
    }
    __syncthreads();

    const bf16_t* ap0 = sA + (wm * 32 + l15) * 64 + half * 16;
    const bf16_t* ap1 = ap0 + 16 * 64;
    const bf16_t* bp0 = sB + (wn * 32 + l15) * 64 + half * 16;
    const bf16_t* bp1 = bp0 + 16 * 64;
#pragma unroll
    for (int kt = 0; kt < 2; ++kt) {
      bf16x16 af0 = *(const bf16x16*)(ap0 + kt * 32);   // 2x ds_load_b128 each
      bf16x16 af1 = *(const bf16x16*)(ap1 + kt * 32);
      bf16x16 bf0 = *(const bf16x16*)(bp0 + kt * 32);
      bf16x16 bf1 = *(const bf16x16*)(bp1 + kt * 32);
      acc00 = __builtin_amdgcn_wmma_f32_16x16x32_bf16(
          false, af0, false, bf0, (short)0, acc00, false, false);
      acc01 = __builtin_amdgcn_wmma_f32_16x16x32_bf16(
          false, af0, false, bf1, (short)0, acc01, false, false);
      acc10 = __builtin_amdgcn_wmma_f32_16x16x32_bf16(
          false, af1, false, bf0, (short)0, acc10, false, false);
      acc11 = __builtin_amdgcn_wmma_f32_16x16x32_bf16(
          false, af1, false, bf1, (short)0, acc11, false, false);
    }
    __syncthreads();
  }

  // Store: C/D f32 layout: VGPR r -> row (mtile*16 + r + half*8), col = l15.
  const int colb = n0 + wn * 32 + l15;
  const float bv0 = bias[colb];
  const float bv1 = bias[colb + 16];
#pragma unroll
  for (int r = 0; r < 8; ++r) {
    int row0 = m0 + wm * 32 + r + half * 8;
    int row1 = row0 + 16;
    float* c0 = C + (size_t)row0 * Nc;
    float* c1 = C + (size_t)row1 * Nc;
    c0[colb]      = acc00[r] + bv0;
    c0[colb + 16] = acc01[r] + bv1;
    c1[colb]      = acc10[r] + bv0;
    c1[colb + 16] = acc11[r] + bv1;
  }
}

// -------------------------- batchnorm (global over rows) ---------------------
__global__ __launch_bounds__(256) void k_bn_stats(
    const float* __restrict__ h, int M, int Nc,
    float* __restrict__ sums, float* __restrict__ sqs) {
  int ch = blockIdx.x;
  float s = 0.0f, q = 0.0f;
  for (int r = threadIdx.x; r < M; r += 256) {
    float v = h[(size_t)r * Nc + ch];
    s += v; q += v * v;
  }
  __shared__ float rs[256], rq[256];
  rs[threadIdx.x] = s; rq[threadIdx.x] = q;
  __syncthreads();
  for (int o = 128; o > 0; o >>= 1) {
    if (threadIdx.x < o) {
      rs[threadIdx.x] += rs[threadIdx.x + o];
      rq[threadIdx.x] += rq[threadIdx.x + o];
    }
    __syncthreads();
  }
  if (threadIdx.x == 0) { sums[ch] = rs[0]; sqs[ch] = rq[0]; }
}

__global__ __launch_bounds__(256) void k_bn_coef(
    const float* __restrict__ sums, const float* __restrict__ sqs,
    const float* __restrict__ gamma, const float* __restrict__ beta,
    int M, int Nc, float* __restrict__ scale, float* __restrict__ shift) {
  int c = blockIdx.x * blockDim.x + threadIdx.x;
  if (c >= Nc) return;
  float inv = 1.0f / (float)M;
  float mu = sums[c] * inv;
  float var = sqs[c] * inv - mu * mu;
  float sc = gamma[c] * rsqrtf(var + BN_EPS);
  scale[c] = sc;
  shift[c] = beta[c] - mu * sc;
}

// Normalize + ReLU + pack to bf16 (input of the next GEMM / maxpool).
__global__ __launch_bounds__(256) void k_bn_relu_pack(
    const float* __restrict__ h, const float* __restrict__ scale,
    const float* __restrict__ shift, size_t total, int Nc,
    bf16_t* __restrict__ out) {
  size_t i = (size_t)blockIdx.x * blockDim.x + threadIdx.x;
  if (i >= total) return;
  int c = (int)(i % (size_t)Nc);
  float v = h[i] * scale[c] + shift[c];
  out[i] = f2bf(v > 0.0f ? v : 0.0f);
}

// -------------------------- max pool over neighborhood -----------------------
// feat[b,s,c] = max_k x[((b*S+s)*K+k)*Nc + c]; outT is (B, Nc, S).
__global__ __launch_bounds__(256) void k_maxpool_bf(
    const bf16_t* __restrict__ x, int S, int K, int Nc,
    float* __restrict__ feat, float* __restrict__ outT) {
  int i = blockIdx.x * blockDim.x + threadIdx.x;
  if (i >= BATCH * S * Nc) return;
  int b = i / (S * Nc);
  int s = (i / Nc) % S;
  int c = i % Nc;
  const bf16_t* base = x + ((size_t)(b * S + s) * K) * Nc + c;
  float m = -INFINITY;
  for (int k = 0; k < K; ++k) m = fmaxf(m, (float)base[(size_t)k * Nc]);
  feat[i] = m;
  outT[(size_t)b * Nc * S + (size_t)c * S + s] = m;
}

// ---------------------------------------------------------------------------
extern "C" void kernel_launch(void* const* d_in, const int* in_sizes, int n_in,
                              void* d_out, int out_size, void* d_ws, size_t ws_size,
                              hipStream_t stream) {
  const float* xyz_in = (const float*)d_in[0];  // (B,3,N)
  const float* pts_in = (const float*)d_in[1];  // (B,3,N)

  const float *Wp[4][3], *Bp[4][3], *Gp[4][3], *Bt[4][3];
  int q = 2;
  for (int st = 0; st < 4; ++st)
    for (int l = 0; l < 3; ++l) {
      Wp[st][l] = (const float*)d_in[q++];
      Bp[st][l] = (const float*)d_in[q++];
      Gp[st][l] = (const float*)d_in[q++];
      Bt[st][l] = (const float*)d_in[q++];
    }

  // Workspace carve-up.
  char* w = (char*)d_ws;
  auto alloc = [&](size_t bytes) -> void* {
    void* p = (void*)w;
    w += (bytes + 255) & ~(size_t)255;
    return p;
  };
  float*  xyzT   = (float*)alloc((size_t)BATCH * NPTS * 3 * 4);
  float*  featA  = (float*)alloc((size_t)BATCH * 1024 * 128 * 4);
  float*  featB  = (float*)alloc((size_t)BATCH * 1024 * 128 * 4);
  float*  nxA    = (float*)alloc((size_t)BATCH * 1024 * 3 * 4);
  float*  nxB    = (float*)alloc((size_t)BATCH * 1024 * 3 * 4);
  int*    fpsidx = (int*)  alloc((size_t)BATCH * 1024 * 4);
  int*    ballix = (int*)  alloc((size_t)BATCH * 1024 * 32 * 4);
  float*  sums   = (float*)alloc(1024 * 4);
  float*  sqs    = (float*)alloc(1024 * 4);
  float*  scale  = (float*)alloc(1024 * 4);
  float*  shift  = (float*)alloc(1024 * 4);
  bf16_t* Gbuf   = (bf16_t*)alloc((size_t)65536 * 192 * 2);   // padded grouped rows
  bf16_t* Xa     = (bf16_t*)alloc((size_t)262144 * 128 * 2);  // bf16 activations
  bf16_t* Xb     = (bf16_t*)alloc((size_t)262144 * 128 * 2);
  float*  Hf     = (float*)alloc((size_t)262144 * 128 * 4);   // f32 GEMM output
  bf16_t* Wbuf   = (bf16_t*)alloc((size_t)1024 * 576 * 2);    // packed weights

  const int   npointA[3]  = {1024, 256, 64};
  const int   nsampleA[3] = {32, 32, 16};
  const float radiusA[3]  = {0.2f, 0.4f, 0.8f};
  const int   mlpA[4][3]  = {{64, 64, 128}, {128, 128, 256},
                             {256, 256, 512}, {512, 512, 1024}};
  const size_t outOff[4]  = {0, 1048576, 1572864, 1835008};
  float* outp = (float*)d_out;
  auto pad64 = [](int x) { return (x + 63) & ~63; };

  // Transpose inputs to (B,n,3).
  {
    int total = BATCH * NPTS;
    k_transpose_b3n<<<(total + 255) / 256, 256, 0, stream>>>(xyz_in, xyzT, NPTS);
    k_transpose_b3n<<<(total + 255) / 256, 256, 0, stream>>>(pts_in, featA, NPTS);
  }

  float* xyzCur  = xyzT;
  float* featCur = featA;
  int n = NPTS, Cprev = 3;

  // Runs one MLP stack (3 layers) on bf16 input `in` (ld = Kld0, rows = M).
  auto run_mlp = [&](int st, int M, const bf16_t* in0, int Cin, int Kld0,
                     bf16_t*& lastOut, int& lastNc) {
    const bf16_t* in = in0;
    int Kld = Kld0, Kc = Cin;
    bf16_t* xb[2] = {Xa, Xb};
    for (int l = 0; l < 3; ++l) {
      const int Nc = mlpA[st][l];
      k_pack_w<<<(Nc * Kld + 255) / 256, 256, 0, stream>>>(Wp[st][l], Nc, Kc, Kld, Wbuf);
      dim3 grid((unsigned)(M / 128), (unsigned)(Nc / 64));
      k_gemm_bf16<<<grid, 256, 0, stream>>>(in, Wbuf, Bp[st][l], Hf, M, Kld, Nc);
      k_bn_stats<<<Nc, 256, 0, stream>>>(Hf, M, Nc, sums, sqs);
      k_bn_coef<<<(Nc + 255) / 256, 256, 0, stream>>>(sums, sqs, Gp[st][l], Bt[st][l],
                                                      M, Nc, scale, shift);
      size_t total = (size_t)M * Nc;
      k_bn_relu_pack<<<(unsigned)((total + 255) / 256), 256, 0, stream>>>(
          Hf, scale, shift, total, Nc, xb[l & 1]);
      in = xb[l & 1];
      Kld = Nc; Kc = Nc;
    }
    lastOut = xb[2 & 1];   // layer 2 wrote xb[0]
    lastNc = mlpA[st][2];
  };

  for (int st = 0; st < 3; ++st) {
    const int S = npointA[st], K = nsampleA[st];
    const float r2 = radiusA[st] * radiusA[st];
    float* nxOut = (st & 1) ? nxB : nxA;

    k_fps<<<BATCH, 256, 0, stream>>>(xyzCur, n, S, fpsidx);
    k_gather_xyz<<<(BATCH * S + 255) / 256, 256, 0, stream>>>(xyzCur, fpsidx, n, S, nxOut);
    k_ball<<<(BATCH * S + 255) / 256, 256, 0, stream>>>(xyzCur, nxOut, n, S, K, r2, ballix);

    const int Cin = 3 + Cprev;
    const int Cld = pad64(Cin);
    const int M = BATCH * S * K;
    k_group_bf<<<(M + 255) / 256, 256, 0, stream>>>(xyzCur, nxOut, featCur,
                                                    n, S, K, Cprev, Cld, ballix, Gbuf);

    bf16_t* lastOut; int lastNc;
    run_mlp(st, M, Gbuf, Cin, Cld, lastOut, lastNc);

    float* featNext = (featCur == featA) ? featB : featA;
    k_maxpool_bf<<<(BATCH * S * lastNc + 255) / 256, 256, 0, stream>>>(
        lastOut, S, K, lastNc, featNext, outp + outOff[st]);

    xyzCur = nxOut; featCur = featNext; n = S; Cprev = lastNc;
  }

  // Stage 3: group_all (n = 64, Cprev = 512).
  {
    const int K = n;                       // 64
    const int M = BATCH * K;               // 512
    const int Cin = 3 + Cprev;             // 515
    const int Cld = pad64(Cin);            // 576
    k_group_all_bf<<<(M + 255) / 256, 256, 0, stream>>>(xyzCur, featCur, n, Cprev, Cld, Gbuf);

    bf16_t* lastOut; int lastNc;
    run_mlp(3, M, Gbuf, Cin, Cld, lastOut, lastNc);

    float* featNext = (featCur == featA) ? featB : featA;
    k_maxpool_bf<<<(BATCH * 1 * lastNc + 255) / 256, 256, 0, stream>>>(
        lastOut, 1, K, lastNc, featNext, outp + outOff[3]);
  }
}
